// GraphSAGE_7224134992217
// MI455X (gfx1250) — matrix operand order repaired and checked by reading, not compiled, not run on hardware
//
#include <hip/hip_runtime.h>

typedef __attribute__((ext_vector_type(16))) _Float16 v16h;
typedef __attribute__((ext_vector_type(8)))  float    v8f;
typedef __attribute__((ext_vector_type(4)))  int      v4i;

#define N_NODES 100000
#define N_EDGES 1600000
#define D 128

#define AS1 __attribute__((address_space(1)))
#define AS3 __attribute__((address_space(3)))

#if defined(__has_builtin)
#if __has_builtin(__builtin_amdgcn_global_load_async_to_lds_b128)
#define HAVE_ASYNC_LDS 1
#endif
#endif

// ---------------------------------------------------------------------------
// Native f32 atomic add (no return value used -> global_atomic_add_f32).
// ---------------------------------------------------------------------------
__device__ __forceinline__ void atomAddF(float* p, float v) {
    __hip_atomic_fetch_add(p, v, __ATOMIC_RELAXED, __HIP_MEMORY_SCOPE_AGENT);
}

// ---------------------------------------------------------------------------
// 16B global -> LDS copy. Async (ASYNCcnt-tracked, no VGPR round trip) when
// the gfx1250 builtin exists; plain load/store fallback otherwise.
// Generic->AS3 conversion: LDS offset lives in the low 32 bits (ISA 10.2).
// Builtin signature wants typed AS pointers: (AS1 v4i*, AS3 v4i*, imm, cpol).
// ---------------------------------------------------------------------------
__device__ __forceinline__ void cp16_g2lds(const void* g, void* l) {
#ifdef HAVE_ASYNC_LDS
    __builtin_amdgcn_global_load_async_to_lds_b128(
        (AS1 v4i*)(unsigned long long)g,
        (AS3 v4i*)(unsigned int)(unsigned long long)l,
        /*imm offset*/0, /*cpol*/0);
#else
    *(uint4*)l = *(const uint4*)g;
#endif
}

__device__ __forceinline__ void wait_async_copies() {
#ifdef HAVE_ASYNC_LDS
#if defined(__has_builtin) && __has_builtin(__builtin_amdgcn_s_wait_asynccnt)
    __builtin_amdgcn_s_wait_asynccnt(0);
#else
    asm volatile("s_wait_asynccnt 0" ::: "memory");
#endif
#endif
}

// ---------------------------------------------------------------------------
// Zero a region of float4s (grid-stride).
// ---------------------------------------------------------------------------
__global__ void __launch_bounds__(256) zero_f4(float4* __restrict__ p, long n4) {
    long i = (long)blockIdx.x * blockDim.x + threadIdx.x;
    long stride = (long)gridDim.x * blockDim.x;
    float4 z; z.x = z.y = z.z = z.w = 0.0f;
    for (; i < n4; i += stride) p[i] = z;
}

// ---------------------------------------------------------------------------
// Pre-pack a 128x128 f32 weight matrix (stored [out,in] row-major; we need
// B = W^T, i.e. B[k][n] = W[n][k]) into f16 WMMA B-fragments.
//
// B-fragment layout used here (contiguous K order; A is built to match):
//   col n = lane & 15 ; lane group g = lane >> 4
//   VGPR v (0..7), half h (0..1): K = g*16 + 2*v + h
// Packed storage: Bp[((ct*4 + kc)*32 + lane)*8 + v]  (uint = 2 packed halves)
// so each lane reads 8 contiguous dwords (32 B) per fragment.
// ---------------------------------------------------------------------------
__global__ void __launch_bounds__(256) pack_weights(const float* __restrict__ W,
                                                    unsigned int* __restrict__ Bp) {
    int idx = blockIdx.x * 256 + threadIdx.x;      // 8*4*32*8 = 8192 entries
    if (idx >= 8192) return;
    int v    = idx & 7;
    int lane = (idx >> 3) & 31;
    int kc   = (idx >> 8) & 3;
    int ct   = idx >> 10;
    int n  = ct * 16 + (lane & 15);
    int k0 = kc * 32 + (lane >> 4) * 16 + 2 * v;
    _Float16 lo = (_Float16)W[n * D + k0];
    _Float16 hi = (_Float16)W[n * D + k0 + 1];
    unsigned short lb = __builtin_bit_cast(unsigned short, lo);
    unsigned short hb = __builtin_bit_cast(unsigned short, hi);
    Bp[idx] = (unsigned)lb | ((unsigned)hb << 16);
}

// ---------------------------------------------------------------------------
// Scatter-sum: one wave per edge. Lane l moves features [4l..4l+3] of
// feat[src] into sum[dst] with native f32 atomics. Lane 0 optionally bumps
// the degree counter. Entirely L2-resident on MI455X (51.2 MB vs 192 MB L2).
// ---------------------------------------------------------------------------
__global__ void __launch_bounds__(256) scatter_sum(const float* __restrict__ feat,
                                                   const int* __restrict__ src,
                                                   const int* __restrict__ dst,
                                                   float* __restrict__ sum,
                                                   float* __restrict__ deg,
                                                   int n_edges, int add_deg) {
    int wave = (int)((blockIdx.x * (unsigned)blockDim.x + threadIdx.x) >> 5);
    int lane = threadIdx.x & 31;
    if (wave >= n_edges) return;
    int s = src[wave];
    int d = dst[wave];
    const float4 val = ((const float4*)(feat + (long)s * D))[lane];
    float* op = sum + (long)d * D + lane * 4;
    atomAddF(op + 0, val.x);
    atomAddF(op + 1, val.y);
    atomAddF(op + 2, val.z);
    atomAddF(op + 3, val.w);
    if (add_deg && lane == 0) atomAddF(deg + d, 1.0f);
}

// ---------------------------------------------------------------------------
// Fused SAGE linear layer:
//   out = act( (sum/deg) @ Wl^T + bl + x @ Wr^T )
// Block = 8 waves; both packed f16 weight matrices (2 x 32 KB) are staged
// into LDS once per block with async global->LDS copies, then every wave
// computes one 16-row tile x all 128 outputs: 4 K-chunks x 16 WMMAs fed by
// ds_load_b128 B-fragments and f32->f16-converted A-fragments (mean folded
// into the agg conversion).
//
// A-fragment layout (16-bit A 16x32, matching packed B): lane L, g = L>>4,
// row M = L&15 ; element e -> K = g*16 + e.
// ---------------------------------------------------------------------------
__device__ __forceinline__ v16h load_bfrag(const unsigned int* p) {
    union { uint4 q[2]; v16h h; } t;
    t.q[0] = ((const uint4*)p)[0];
    t.q[1] = ((const uint4*)p)[1];
    return t.h;
}

__global__ void __launch_bounds__(256)
sage_gemm(const float* __restrict__ ssum, const float* __restrict__ deg,
          const float* __restrict__ xin,
          const unsigned int* __restrict__ Bl, const unsigned int* __restrict__ Br,
          const float* __restrict__ bias, float* __restrict__ out, int relu) {
    __shared__ unsigned int lds_w[2][8192];        // 64 KB: [0]=Wl frags, [1]=Wr frags

    // ---- Stage both weight matrices into LDS (all 256 threads, pre-return).
    {
        int t = threadIdx.x;
#pragma unroll
        for (int j = 0; j < 8; ++j) {
            int q = j * 256 + t;                   // uint4 index, 0..2047
            cp16_g2lds(Bl + q * 4, &lds_w[0][q * 4]);
            cp16_g2lds(Br + q * 4, &lds_w[1][q * 4]);
        }
        wait_async_copies();
        __syncthreads();
    }

    int wv   = threadIdx.x >> 5;
    int lane = threadIdx.x & 31;
    int tile = blockIdx.x * 8 + wv;                 // 6250 tiles total
    if (tile * 16 >= N_NODES) return;               // uniform per wave: EXEC all-ones
    int row0 = tile * 16;
    int r    = row0 + (lane & 15);
    int g    = lane >> 4;

    float inv = 1.0f / fmaxf(deg[r], 1.0f);

    v8f acc[8] = {};                                // 8 col tiles x 16x16 f32

#pragma unroll
    for (int kc = 0; kc < 4; ++kc) {
        const int kbase = kc * 32 + g * 16;
        const float* a1p = ssum + (long)r * D + kbase;
        const float* a2p = xin  + (long)r * D + kbase;
        v16h a1, a2;
#pragma unroll
        for (int e = 0; e < 8; ++e) {
            a1[e]     = (_Float16)(a1p[e]      * inv);
            a1[e + 8] = (_Float16)(a1p[e + 8] * inv);
            a2[e]     = (_Float16)a2p[e];
            a2[e + 8] = (_Float16)a2p[e + 8];
        }
#pragma unroll
        for (int ct = 0; ct < 8; ++ct) {
            const int fidx = ((ct * 4 + kc) * 32 + lane) * 8;
            v16h bl = load_bfrag(&lds_w[0][fidx]);
            acc[ct] = __builtin_amdgcn_wmma_f32_16x16x32_f16(
                false, a1, false, bl, (short)0, acc[ct], false, false);
            v16h br = load_bfrag(&lds_w[1][fidx]);
            acc[ct] = __builtin_amdgcn_wmma_f32_16x16x32_f16(
                false, a2, false, br, (short)0, acc[ct], false, false);
        }
    }

    // Epilogue: bias, optional ReLU, store. C/D layout: VGPR i -> row g*8+i,
    // col = ct*16 + (lane&15).
#pragma unroll
    for (int ct = 0; ct < 8; ++ct) {
        int n = ct * 16 + (lane & 15);
        float b = bias[n];
#pragma unroll
        for (int i = 0; i < 8; ++i) {
            float v = acc[ct][i] + b;
            if (relu) v = fmaxf(v, 0.0f);
            out[(long)(row0 + g * 8 + i) * D + n] = v;
        }
    }
}

// ---------------------------------------------------------------------------
// Host-side orchestration.
// Workspace layout (bytes):
//   deg   : N_NODES*4                =    400,000
//   sum1  : N_NODES*D*4              = 51,200,000
//   sum2  : N_NODES*D*4              = 51,200,000
//   h     : N_NODES*D*4              = 51,200,000
//   4x packed weights (8192 u32)     =    131,072
// total ~154.1 MB
// ---------------------------------------------------------------------------
extern "C" void kernel_launch(void* const* d_in, const int* in_sizes, int n_in,
                              void* d_out, int out_size, void* d_ws, size_t ws_size,
                              hipStream_t stream) {
    (void)in_sizes; (void)n_in; (void)out_size; (void)ws_size;

    const float* x   = (const float*)d_in[0];
    const int*  edge = (const int*)d_in[1];
    const int*  src  = edge;
    const int*  dst  = edge + N_EDGES;
    const float* W1l = (const float*)d_in[2];
    const float* b1l = (const float*)d_in[3];
    const float* W1r = (const float*)d_in[4];
    const float* W2l = (const float*)d_in[5];
    const float* b2l = (const float*)d_in[6];
    const float* W2r = (const float*)d_in[7];
    float* out = (float*)d_out;

    char* ws = (char*)d_ws;
    float* deg  = (float*)ws;                 ws += (size_t)N_NODES * 4;
    float* sum1 = (float*)ws;                 ws += (size_t)N_NODES * D * 4;
    float* sum2 = (float*)ws;                 ws += (size_t)N_NODES * D * 4;
    float* hbuf = (float*)ws;                 ws += (size_t)N_NODES * D * 4;
    unsigned int* W1l_pk = (unsigned int*)ws; ws += 8192 * 4;
    unsigned int* W1r_pk = (unsigned int*)ws; ws += 8192 * 4;
    unsigned int* W2l_pk = (unsigned int*)ws; ws += 8192 * 4;
    unsigned int* W2r_pk = (unsigned int*)ws; ws += 8192 * 4;

    // 1) Pack weights into f16 WMMA B-fragment layout (tiny).
    pack_weights<<<32, 256, 0, stream>>>(W1l, W1l_pk);
    pack_weights<<<32, 256, 0, stream>>>(W1r, W1r_pk);
    pack_weights<<<32, 256, 0, stream>>>(W2l, W2l_pk);
    pack_weights<<<32, 256, 0, stream>>>(W2r, W2r_pk);

    // 2) Zero deg + sum1 + sum2 (contiguous region).
    long zero_floats = (long)N_NODES + 2L * N_NODES * D;   // 25,700,000
    zero_f4<<<2048, 256, 0, stream>>>((float4*)deg, zero_floats / 4);

    // 3) Layer 1 scatter-mean numerators + degrees.
    scatter_sum<<<N_EDGES / 8, 256, 0, stream>>>(x, src, dst, sum1, deg, N_EDGES, 1);

    // 4) h = relu(mean1 @ W1l^T + b1l + x @ W1r^T)
    int gemm_blocks = (N_NODES / 16 + 7) / 8;              // 782
    sage_gemm<<<gemm_blocks, 256, 0, stream>>>(sum1, deg, x, W1l_pk, W1r_pk, b1l, hbuf, 1);

    // 5) Layer 2 scatter (degrees reused).
    scatter_sum<<<N_EDGES / 8, 256, 0, stream>>>(hbuf, src, dst, sum2, deg, N_EDGES, 0);

    // 6) out = mean2 @ W2l^T + b2l + h @ W2r^T
    sage_gemm<<<gemm_blocks, 256, 0, stream>>>(sum2, deg, hbuf, W2l_pk, W2r_pk, b2l, out, 0);
}
